// GatedScorer_25666724561374
// MI455X (gfx1250) — matrix operand order repaired and checked by reading, not compiled, not run on hardware
//
#include <hip/hip_runtime.h>
#include <math.h>

// ---------------------------------------------------------------------------
// GatedScorer for MI455X (gfx1250, wave32, WMMA + TDM).
//
//   s(b,q,k) = sum_h  U * gelu(V) * Wo[h] + bo
//   U = rstd*( Au(b,q,h) + Bu(b,k,h) + P_u(b,q,k,h) - mu*su[h] ) + cu[h]
//   LayerNorm folded into weights; only the (q .* k) @ WuP'/WvP' per-pair
//   GEMM remains (137 GFLOP) -> bf16 WMMA 16x16x32, f32 accumulate.
//   k-row tiles staged into LDS by the Tensor Data Mover (6-arg builtin).
// ---------------------------------------------------------------------------

#define B_   2
#define QL_  256
#define KL_  256
#define QD_  1024
#define H_   512
#define C3_  1536
#define EPS_ 1e-5f

typedef __attribute__((ext_vector_type(2)))  float   v2f;
typedef __attribute__((ext_vector_type(8)))  float   v8f;
typedef __attribute__((ext_vector_type(4)))  __bf16  v4bf;
typedef __attribute__((ext_vector_type(8)))  __bf16  v8bf;
typedef __attribute__((ext_vector_type(16))) __bf16  v16bf;

#if defined(__has_builtin)
#  if __has_builtin(__builtin_amdgcn_wmma_f32_16x16x4_f32)
#    define HAVE_WMMA_F32K4 1
#  else
#    define HAVE_WMMA_F32K4 0
#  endif
#  if __has_builtin(__builtin_amdgcn_tensor_load_to_lds) && \
      __has_builtin(__builtin_amdgcn_s_wait_tensorcnt)
#    define HAVE_TDM 1
#  else
#    define HAVE_TDM 0
#  endif
#else
#  define HAVE_WMMA_F32K4 0
#  define HAVE_TDM 0
#endif

static __device__ inline v16bf ld_b16x16(const __bf16* p0, const __bf16* p1) {
  v8bf lo = *(const v8bf*)p0;   // 16B -> ds_load_b128 / global_load_b128
  v8bf hi = *(const v8bf*)p1;
  v16bf r;
#pragma unroll
  for (int i = 0; i < 8; ++i) { r[i] = lo[i]; r[8 + i] = hi[i]; }
  return r;
}

#if HAVE_TDM
typedef __attribute__((ext_vector_type(4))) unsigned int u32x4;
typedef __attribute__((ext_vector_type(8))) int          i32x8;
typedef __attribute__((ext_vector_type(4))) int          i32x4;

// TDM 2D tile load: rows x cols f32 from global (row stride = cols) into LDS.
// D# per CDNA5 ISA ch.8: group0 = {flags, lds_addr, global_addr, type=2},
// group1 = {data_size=4B, tensor dims, tile dims, dim0 stride}.
// This toolchain exposes the 6-arg builtin (extra i32x8 group before cpol).
static __device__ inline void tdm_load_rows_f32(void* ldsDst, const float* gsrc,
                                                unsigned rows, unsigned cols) {
  const unsigned long long ga = (unsigned long long)(uintptr_t)gsrc;
  const unsigned lds = (unsigned)(uintptr_t)ldsDst;  // low 32 bits = LDS offset
  u32x4 g0;
  g0[0] = 1u;                                   // count=1, user mode
  g0[1] = lds;                                  // lds_addr
  g0[2] = (unsigned)ga;                         // global_addr[31:0]
  g0[3] = (unsigned)((ga >> 32) & 0x01FFFFFFu)  // global_addr[56:32]
          | 0x80000000u;                        // type=2 ("image") in [127:126]
  i32x8 g1;
  g1[0] = (int)(2u << 16);                      // wg_mask=0, data_size=2 (4B)
  g1[1] = (int)((cols & 0xFFFFu) << 16);        // barrier_addr=0 | tensor_dim0.lo
  g1[2] = (int)((cols >> 16) | (rows << 16));   // tensor_dim0.hi | tensor_dim1.lo
  g1[3] = (int)((rows >> 16) | (cols << 16));   // tensor_dim1.hi | tile_dim0
  g1[4] = (int)(rows & 0xFFFFu);                // tile_dim1 | tile_dim2=0
  g1[5] = (int)cols;                            // tensor_dim0_stride[31:0]
  g1[6] = 0;                                    // stride[47:32] | dim1_stride.lo
  g1[7] = 0;
  i32x4 g2 = {0, 0, 0, 0};
  i32x4 g3 = {0, 0, 0, 0};
  i32x8 g4 = {0, 0, 0, 0, 0, 0, 0, 0};
  __builtin_amdgcn_tensor_load_to_lds(g0, g1, g2, g3, g4, 0);
}
#endif

// ---------------------------------------------------------------------------
// 1) Fold LayerNorm gamma into Wu/Wv; emit f32 q/k blocks, bf16 pair block,
//    and the per-h constants su,cu,sv,cv.
// ---------------------------------------------------------------------------
__global__ __launch_bounds__(256)
void prep_weights(const float* __restrict__ Wu, const float* __restrict__ Wv,
                  const float* __restrict__ lnw, const float* __restrict__ lnb,
                  const float* __restrict__ bu,  const float* __restrict__ bv,
                  float* __restrict__ WuQ, float* __restrict__ WuK,
                  float* __restrict__ WvQ, float* __restrict__ WvK,
                  __bf16* __restrict__ WuP, __bf16* __restrict__ WvP,
                  float* __restrict__ su, float* __restrict__ cu,
                  float* __restrict__ sv, float* __restrict__ cv) {
  __shared__ float r0[256], r1[256], r2[256], r3[256];
  const int h = blockIdx.x, t = threadIdx.x;
  float su_ = 0.f, cu_ = 0.f, sv_ = 0.f, cv_ = 0.f;
  for (int c = t; c < C3_; c += 256) {
    const float lw = lnw[c], lb = lnb[c];
    const float wu = Wu[h * C3_ + c], wv = Wv[h * C3_ + c];
    su_ += wu * lw; cu_ += wu * lb; sv_ += wv * lw; cv_ += wv * lb;
    const float swu = wu * lw, swv = wv * lw;
    if (c < H_)            { WuQ[h * H_ + c] = swu;            WvQ[h * H_ + c] = swv; }
    else if (c < 2 * H_)   { WuK[h * H_ + c - H_] = swu;       WvK[h * H_ + c - H_] = swv; }
    else                   { WuP[h * H_ + c - 2 * H_] = (__bf16)swu;
                             WvP[h * H_ + c - 2 * H_] = (__bf16)swv; }
  }
  r0[t] = su_; r1[t] = cu_; r2[t] = sv_; r3[t] = cv_;
  __syncthreads();
  for (int s = 128; s > 0; s >>= 1) {
    if (t < s) { r0[t] += r0[t + s]; r1[t] += r1[t + s];
                 r2[t] += r2[t + s]; r3[t] += r3[t + s]; }
    __syncthreads();
  }
  if (t == 0) { su[h] = r0[0]; cu[h] = r1[0] + bu[h];
                sv[h] = r2[0]; cv[h] = r3[0] + bv[h]; }
}

// ---------------------------------------------------------------------------
// 2) Generic f32 GEMM  C[M,N] = A[M,K] @ W[N,K]^T  via v_wmma_f32_16x16x4_f32.
// ---------------------------------------------------------------------------
__global__ __launch_bounds__(256)
void gemm_f32(const float* __restrict__ A, const float* __restrict__ W,
              float* __restrict__ C, int M, int N, int K) {
  const int wave = threadIdx.x >> 5, lane = threadIdx.x & 31;
  const int l = lane & 15, half = lane >> 4;
  const int gid = blockIdx.x * 8 + wave;
  const int ntiles = N >> 4;
  const int tm = (gid / ntiles) << 4, tn = (gid % ntiles) << 4;
  if (tm >= M) return;  // wave-uniform
  v8f acc = {};
#if HAVE_WMMA_F32K4
  const float* ap = A + (size_t)(tm + l) * K + 2 * half;
  const float* bp = W + (size_t)(tn + l) * K + 2 * half;
  for (int k = 0; k < K; k += 4) {
    v2f av = *(const v2f*)(ap + k);
    v2f bv = *(const v2f*)(bp + k);
    acc = __builtin_amdgcn_wmma_f32_16x16x4_f32(false, av, false, bv,
                                                (short)0, acc, false, false);
  }
#else
  for (int k = 0; k < K; ++k) {
    const float bvv = W[(size_t)(tn + l) * K + k];
#pragma unroll
    for (int i = 0; i < 8; ++i)
      acc[i] += A[(size_t)(tm + 8 * half + i) * K + k] * bvv;
  }
#endif
#pragma unroll
  for (int i = 0; i < 8; ++i)
    C[(size_t)(tm + 8 * half + i) * N + tn + l] = acc[i];
}

// ---------------------------------------------------------------------------
// 3) Per-row sums / sums of squares of qh and kh (for LayerNorm stats).
// ---------------------------------------------------------------------------
__global__ __launch_bounds__(128)
void row_stats(const float* __restrict__ qh, const float* __restrict__ kh,
               float* __restrict__ Sq, float* __restrict__ Q2,
               float* __restrict__ Sk, float* __restrict__ K2) {
  __shared__ float r0[128], r1[128];
  const int r = blockIdx.x, t = threadIdx.x;
  const float* src = (r < B_ * QL_) ? (qh + (size_t)r * H_)
                                    : (kh + (size_t)(r - B_ * QL_) * H_);
  float s = 0.f, s2 = 0.f;
  for (int c = t; c < H_; c += 128) { const float v = src[c]; s += v; s2 += v * v; }
  r0[t] = s; r1[t] = s2;
  __syncthreads();
  for (int st = 64; st > 0; st >>= 1) {
    if (t < st) { r0[t] += r0[t + st]; r1[t] += r1[t + st]; }
    __syncthreads();
  }
  if (t == 0) {
    if (r < B_ * QL_) { Sq[r] = r0[0]; Q2[r] = r1[0]; }
    else              { Sk[r - B_ * QL_] = r0[0]; K2[r - B_ * QL_] = r1[0]; }
  }
}

// ---------------------------------------------------------------------------
// 4) Main fused kernel. One block = one (b,q) x 32 k-rows x all 512 h.
// ---------------------------------------------------------------------------
#define AST 520  // LDS a-tile row stride in bf16 (pad 8 -> 16B bank rotation)

// per-row: products q.*k -> bf16 a-tile + exact f32 Sp / P2 (wave32 reduce)
static __device__ inline void make_pair_row(const float* __restrict__ qrow,
                                            const float* __restrict__ krow,
                                            int j, int lane,
                                            __bf16* aT, float* SpA, float* P2A) {
  float sp = 0.f, p2 = 0.f;
#pragma unroll
  for (int i = 0; i < 4; ++i) {
    const int c = lane * 4 + i * 128;
    const float4 qv = *(const float4*)(qrow + c);
    const float4 kv = *(const float4*)(krow + c);
    const float px = qv.x * kv.x, py = qv.y * kv.y;
    const float pz = qv.z * kv.z, pw = qv.w * kv.w;
    sp += px + py + pz + pw;
    p2 += px * px + py * py + pz * pz + pw * pw;
    v4bf pb = { (__bf16)px, (__bf16)py, (__bf16)pz, (__bf16)pw };
    *(v4bf*)(&aT[j * AST + c]) = pb;
  }
  sp += __shfl_xor(sp, 1);  p2 += __shfl_xor(p2, 1);
  sp += __shfl_xor(sp, 2);  p2 += __shfl_xor(p2, 2);
  sp += __shfl_xor(sp, 4);  p2 += __shfl_xor(p2, 4);
  sp += __shfl_xor(sp, 8);  p2 += __shfl_xor(p2, 8);
  sp += __shfl_xor(sp, 16); p2 += __shfl_xor(p2, 16);
  if (lane == 0) { SpA[j] = sp; P2A[j] = p2; }
}

__global__ __launch_bounds__(256)
void pair_kernel(const float* __restrict__ qh, const float* __restrict__ kh,
                 const float* __restrict__ Au, const float* __restrict__ Bu,
                 const float* __restrict__ Av, const float* __restrict__ Bv,
                 const __bf16* __restrict__ WuP, const __bf16* __restrict__ WvP,
                 const float* __restrict__ su, const float* __restrict__ cu,
                 const float* __restrict__ sv, const float* __restrict__ cv,
                 const float* __restrict__ Sq, const float* __restrict__ Q2,
                 const float* __restrict__ Sk, const float* __restrict__ K2,
                 const float* __restrict__ Wo, const float* __restrict__ bo,
                 float* __restrict__ out) {
  __shared__ __bf16 aT[32 * AST];            // 33280 B
  __shared__ float SpA[32], P2A[32];
  __shared__ float muA[32], rsA[32];
  __shared__ float sPart[32][8];
#if HAVE_TDM
  __shared__ float kLds[8 * H_];             // 16 KB TDM staging chunk
#endif

  const int tid = threadIdx.x;
  const int wave = tid >> 5, lane = tid & 31;
  const int l = lane & 15, half = lane >> 4;
  const int bq = blockIdx.x >> 3;            // b*QL + q  (0..511)
  const int b = bq >> 8;
  const int kbase = (blockIdx.x & 7) << 5;   // 0,32,...,224

  const float* qrow = qh + (size_t)bq * H_;
  const float* kb0  = kh + (size_t)(b * KL_ + kbase) * H_;

  // ---- Phase A: stage k rows (TDM DMA -> LDS), build bf16 a-tile + stats --
#if HAVE_TDM
  for (int cc = 0; cc < 4; ++cc) {
    if (wave == 0) {                         // one TDM op per chunk, wave 0
      tdm_load_rows_f32(kLds, kb0 + (size_t)cc * 8 * H_, 8u, (unsigned)H_);
      __builtin_amdgcn_s_wait_tensorcnt(0);
    }
    __syncthreads();                         // LDS chunk visible to all waves
    make_pair_row(qrow, (const float*)&kLds[wave * H_],
                  cc * 8 + wave, lane, aT, SpA, P2A);
    __syncthreads();                         // before chunk buffer reuse
  }
#else
  for (int cc = 0; cc < 4; ++cc) {
    const int j = cc * 8 + wave;
    make_pair_row(qrow, kb0 + (size_t)j * H_, j, lane, aT, SpA, P2A);
  }
  __syncthreads();
#endif

  if (tid < 32) {
    const int bk = b * KL_ + kbase + tid;
    const float mu  = (Sq[bq] + Sk[bk] + SpA[tid]) * (1.0f / 1536.0f);
    const float var = (Q2[bq] + K2[bk] + P2A[tid]) * (1.0f / 1536.0f) - mu * mu;
    muA[tid] = mu;
    rsA[tid] = rsqrtf(var + EPS_);
  }
  __syncthreads();

  // ---- Phase B: P_u / P_v = a @ WuP'^T / WvP'^T with bf16 WMMA ----
  const int n0w = wave << 6;  // 64 columns per wave

  v8f accU[2][4], accV[2][4];
  const v8f vzero = {};
#pragma unroll
  for (int mt = 0; mt < 2; ++mt)
#pragma unroll
    for (int nt = 0; nt < 4; ++nt) { accU[mt][nt] = vzero; accV[mt][nt] = vzero; }

  for (int kk = 0; kk < H_; kk += 32) {
    v16bf afr[2];
#pragma unroll
    for (int mt = 0; mt < 2; ++mt) {
      // 16-bit A 16x32 layout: row = l; VGPR0-3 <- K[half*8..+8),
      // VGPR4-7 <- K[16+half*8..+8)  (two ds_load_b128)
      const __bf16* base = &aT[(mt * 16 + l) * AST + kk + half * 8];
      afr[mt] = ld_b16x16(base, base + 16);
    }
#pragma unroll
    for (int nt = 0; nt < 4; ++nt) {
      const int n = n0w + nt * 16 + l;
      // B 32x16: col = l; VGPR0-3 <- K[half*8..+8), VGPR4-7 <- K[16+half*8..+8)
      const __bf16* bup = WuP + (size_t)n * H_ + kk + half * 8;
      const v16bf bfrU = ld_b16x16(bup, bup + 16);
      const __bf16* bvp = WvP + (size_t)n * H_ + kk + half * 8;
      const v16bf bfrV = ld_b16x16(bvp, bvp + 16);
      if (kk + 32 < H_) {                    // warm L0 for next k-step
        __builtin_prefetch((const void*)(bup + 32), 0, 3);
        __builtin_prefetch((const void*)(bvp + 32), 0, 3);
      }
#pragma unroll
      for (int mt = 0; mt < 2; ++mt) {
        accU[mt][nt] = __builtin_amdgcn_wmma_f32_16x16x32_bf16(
            false, afr[mt], false, bfrU, (short)0, accU[mt][nt], false, false);
        accV[mt][nt] = __builtin_amdgcn_wmma_f32_16x16x32_bf16(
            false, afr[mt], false, bfrV, (short)0, accV[mt][nt], false, false);
      }
    }
  }

  // ---- Phase C: epilogue — LayerNorm affine, GELU gate, dot with Wo ----
  float auv[4], avv[4], suv[4], svv[4], cuv[4], cvv[4], wov[4];
#pragma unroll
  for (int nt = 0; nt < 4; ++nt) {
    const int n = n0w + nt * 16 + l;
    auv[nt] = Au[(size_t)bq * H_ + n];
    avv[nt] = Av[(size_t)bq * H_ + n];
    suv[nt] = su[n]; svv[nt] = sv[n];
    cuv[nt] = cu[n]; cvv[nt] = cv[n];
    wov[nt] = Wo[n];
  }
  const int bkr0 = b * KL_ + kbase;
#pragma unroll
  for (int mt = 0; mt < 2; ++mt) {
#pragma unroll
    for (int i = 0; i < 8; ++i) {
      const int m = mt * 16 + half * 8 + i;   // C-frag row mapping
      const float mu = muA[m], rs = rsA[m];
      const float* BuR = Bu + (size_t)(bkr0 + m) * H_;
      const float* BvR = Bv + (size_t)(bkr0 + m) * H_;
      float acc = 0.f;
#pragma unroll
      for (int nt = 0; nt < 4; ++nt) {
        const int n = n0w + nt * 16 + l;
        const float u  = rs * (accU[mt][nt][i] + auv[nt] + BuR[n] - mu * suv[nt]) + cuv[nt];
        const float vv = rs * (accV[mt][nt][i] + avv[nt] + BvR[n] - mu * svv[nt]) + cvv[nt];
        const float g = 0.5f * vv * (1.0f + erff(vv * 0.70710678118654752f));
        acc += u * g * wov[nt];
      }
      // fixed-order reduce across the 16 lanes of each half (deterministic)
      acc += __shfl_xor(acc, 1);
      acc += __shfl_xor(acc, 2);
      acc += __shfl_xor(acc, 4);
      acc += __shfl_xor(acc, 8);
      if (l == 0) sPart[m][wave] = acc;
    }
  }
  __syncthreads();
  if (tid < 32) {
    float s = 0.f;
#pragma unroll
    for (int w = 0; w < 8; ++w) s += sPart[tid][w];
    out[(size_t)bq * KL_ + kbase + tid] = s + bo[0];
  }
}

// ---------------------------------------------------------------------------
// Host launcher
// ---------------------------------------------------------------------------
extern "C" void kernel_launch(void* const* d_in, const int* in_sizes, int n_in,
                              void* d_out, int out_size, void* d_ws, size_t ws_size,
                              hipStream_t stream) {
  (void)in_sizes; (void)n_in; (void)out_size; (void)ws_size;
  const float* Q   = (const float*)d_in[0];
  const float* Kin = (const float*)d_in[1];
  const float* Wq  = (const float*)d_in[2];
  const float* Wk  = (const float*)d_in[3];
  const float* lnw = (const float*)d_in[4];
  const float* lnb = (const float*)d_in[5];
  const float* Wu  = (const float*)d_in[6];
  const float* bu  = (const float*)d_in[7];
  const float* Wv  = (const float*)d_in[8];
  const float* bv  = (const float*)d_in[9];
  const float* Wo  = (const float*)d_in[10];
  const float* bo  = (const float*)d_in[11];
  float* out = (float*)d_out;

  // workspace carve (~11.5 MB)
  float* ws = (float*)d_ws;
  const size_t HH = (size_t)H_ * H_;          // 262144
  const size_t RQ = (size_t)B_ * QL_ * H_;    // 262144
  const size_t RK = (size_t)B_ * KL_ * H_;    // 262144
  float* qh  = ws;
  float* kh  = qh + RQ;
  float* WuQ = kh + RK;
  float* WuK = WuQ + HH;
  float* WvQ = WuK + HH;
  float* WvK = WvQ + HH;
  float* Au  = WvK + HH;
  float* Bu  = Au + RQ;
  float* Av  = Bu + RK;
  float* Bv  = Av + RQ;
  float* su  = Bv + RK;
  float* cu  = su + H_;
  float* sv  = cu + H_;
  float* cv  = sv + H_;
  float* Sq  = cv + H_;
  float* Q2  = Sq + B_ * QL_;
  float* Sk  = Q2 + B_ * QL_;
  float* K2  = Sk + B_ * KL_;
  __bf16* WuP = (__bf16*)(K2 + B_ * KL_);
  __bf16* WvP = WuP + HH;

  prep_weights<<<H_, 256, 0, stream>>>(Wu, Wv, lnw, lnb, bu, bv,
                                       WuQ, WuK, WvQ, WvK, WuP, WvP,
                                       su, cu, sv, cv);

#define GEMM_LAUNCH(Ap, Wp, Cp, M, N, Kd)                                     \
  gemm_f32<<<((M) / 16) * ((N) / 16) / 8, 256, 0, stream>>>(Ap, Wp, Cp, M, N, Kd)

  GEMM_LAUNCH(Q,   Wq,  qh, B_ * QL_, H_, QD_);   // q projection
  GEMM_LAUNCH(Kin, Wk,  kh, B_ * KL_, H_, QD_);   // k projection
  row_stats<<<B_ * (QL_ + KL_), 128, 0, stream>>>(qh, kh, Sq, Q2, Sk, K2);
  GEMM_LAUNCH(qh, WuQ, Au, B_ * QL_, H_, H_);     // rank-reduced q-parts
  GEMM_LAUNCH(kh, WuK, Bu, B_ * KL_, H_, H_);
  GEMM_LAUNCH(qh, WvQ, Av, B_ * QL_, H_, H_);
  GEMM_LAUNCH(kh, WvK, Bv, B_ * KL_, H_, H_);
#undef GEMM_LAUNCH

  pair_kernel<<<B_ * QL_ * (KL_ / 32), 256, 0, stream>>>(
      qh, kh, Au, Bu, Av, Bv, WuP, WvP, su, cu, sv, cv,
      Sq, Q2, Sk, K2, Wo, bo, out);
}
